// DiscoveryNet_16381005267361
// MI455X (gfx1250) — matrix operand order repaired and checked by reading, not compiled, not run on hardware
//
#include <hip/hip_runtime.h>
#include <hip/hip_bf16.h>

typedef __attribute__((ext_vector_type(16))) _Float16 v16h;
typedef __attribute__((ext_vector_type(8)))  _Float16 v8h;
typedef __attribute__((ext_vector_type(8)))  float    v8f;

#define NPTS   512
#define NM1    511
#define HDIM   128
#define PAIRS  (NPTS * NM1)          // 261632, divisible by 16
#define TILES_PER_B (PAIRS / 16)     // 16352
#define NBATCH 8
#define NTILES (NBATCH * TILES_PER_B)
#define PADK   136                   // 128 + 8 halves pad -> 272B row stride, conflict-free

__global__ void disc_init_out(float* out, const float* b3) {
    int i = threadIdx.x;
    if (i < NBATCH) out[i] = 0.5f * (float)PAIRS * b3[0];
}

// Build one 16x32 f16 A-fragment (ISA layout) = silu(feats @ W1 + b1) for this
// wave's K-slice. Staged so the 16 independent TRANS chains pipeline.
__device__ __forceinline__ v16h build_a_frag(const float4* __restrict__ sL1,
                                             int kb, float fx, float fy, float fz) {
    float z[16];
    #pragma unroll
    for (int e = 0; e < 8; ++e) {
        float4 w  = sL1[kb + e];
        z[e]      = fx * w.x  + fy * w.y  + fz * w.z  + w.w;
        float4 w2 = sL1[kb + 16 + e];
        z[8 + e]  = fx * w2.x + fy * w2.y + fz * w2.z + w2.w;
    }
    float s[16];
    #pragma unroll
    for (int e = 0; e < 16; ++e) s[e] = __expf(-z[e]);
    #pragma unroll
    for (int e = 0; e < 16; ++e) s[e] = __builtin_amdgcn_rcpf(1.0f + s[e]);
    v16h a;
    #pragma unroll
    for (int e = 0; e < 16; ++e) a[e] = (_Float16)(z[e] * s[e]);
    return a;
}

__device__ __forceinline__ v16h load_b_frag(const _Float16* bp) {
    v8h blo = *(const v8h*)(bp);
    v8h bhi = *(const v8h*)(bp + 16);
    v16h bf;
    #pragma unroll
    for (int e = 0; e < 8; ++e) { bf[e] = blo[e]; bf[8 + e] = bhi[e]; }
    return bf;
}

__global__ __launch_bounds__(256, 6)
void disc_pair_mlp(const float* __restrict__ pos,
                   const float* __restrict__ W1, const float* __restrict__ b1,
                   const float* __restrict__ W2, const float* __restrict__ b2,
                   const float* __restrict__ W3,
                   float* __restrict__ out) {
    __shared__ _Float16 sW2T[HDIM * PADK];   // W2T[n][k] = W2[k][n], f16, padded rows
    __shared__ float4   sL1[HDIM];           // {W1[0][k], W1[1][k], W1[2][k], b1[k]}
    __shared__ float    sB2[HDIM];
    __shared__ float    sW3[HDIM];

    const int tid = threadIdx.x;

    // ---- one-time per-block weight staging / conversion ----
    for (int idx = tid; idx < HDIM * HDIM; idx += 256) {
        int n = idx >> 7, k = idx & 127;
        sW2T[n * PADK + k] = (_Float16)W2[k * HDIM + n];
    }
    for (int k = tid; k < HDIM; k += 256) {
        sL1[k] = make_float4(W1[k], W1[HDIM + k], W1[2 * HDIM + k], b1[k]);
        sB2[k] = b2[k];
        sW3[k] = W3[k];
    }
    __syncthreads();

    const int lane    = tid & 31;
    const int waveId  = (blockIdx.x * (blockDim.x >> 5)) + (tid >> 5);
    const int nWaves  = gridDim.x * (blockDim.x >> 5);
    const int row     = lane & 15;            // pair row within tile / output column
    const int kHalfA  = (lane >> 4) << 3;     // A-frag K half-offset: 0 or 8
    const int kHalfB  = (lane >> 4) << 3;     // B-frag K half-offset: 0 or 8

    #pragma unroll 1
    for (int t = waveId; t < NTILES; t += nWaves) {
        const int b  = t / TILES_PER_B;
        const int tb = t - b * TILES_PER_B;

        // ---- per-lane pair features (row = lane&15; lane halves duplicate) ----
        const int p = tb * 16 + row;
        const int i = p / NM1;
        const int q = p - i * NM1;
        const int j = q + (q >= i ? 1 : 0);
        const float* pb = pos + (size_t)b * NPTS * 3;
        const float dx = pb[i * 3 + 0] - pb[j * 3 + 0];
        const float dy = pb[i * 3 + 1] - pb[j * 3 + 1];
        const float dz = pb[i * 3 + 2] - pb[j * 3 + 2];
        float r = fmaxf(sqrtf(dx * dx + dy * dy + dz * dz), 0.05f);
        const float fx = r;
        const float fy = 1.0f / r;
        const float fz = fy * fy;

        // ---- layer1: 4 A-fragments (full 16x128 activation, ISA layout) ----
        const v16h a0 = build_a_frag(sL1, 0 * 32 + kHalfA, fx, fy, fz);
        const v16h a1 = build_a_frag(sL1, 1 * 32 + kHalfA, fx, fy, fz);
        const v16h a2 = build_a_frag(sL1, 2 * 32 + kHalfA, fx, fy, fz);
        const v16h a3 = build_a_frag(sL1, 3 * 32 + kHalfA, fx, fy, fz);

        float acc = 0.0f;

        // ---- layer2 GEMM, nt ROLLED so B-fragments can't be hoisted/spilled ----
        #pragma unroll 1
        for (int nt = 0; nt < 8; ++nt) {
            const _Float16* bbase = &sW2T[(nt * 16 + row) * PADK + kHalfB];

            // batch-issue all 8 ds_load_b128 so LDS latency overlaps the WMMAs
            const v16h bf0 = load_b_frag(bbase + 0 * 32);
            const v16h bf1 = load_b_frag(bbase + 1 * 32);
            const v16h bf2 = load_b_frag(bbase + 2 * 32);
            const v16h bf3 = load_b_frag(bbase + 3 * 32);

            v8f c0, c1;
            #pragma unroll
            for (int v = 0; v < 8; ++v) { c0[v] = 0.0f; c1[v] = 0.0f; }

            // two independent accumulator chains -> 2x matrix-pipe ILP
            c0 = __builtin_amdgcn_wmma_f32_16x16x32_f16(false, a0, false, bf0, (short)0, c0, false, false);
            c1 = __builtin_amdgcn_wmma_f32_16x16x32_f16(false, a1, false, bf1, (short)0, c1, false, false);
            c0 = __builtin_amdgcn_wmma_f32_16x16x32_f16(false, a2, false, bf2, (short)0, c0, false, false);
            c1 = __builtin_amdgcn_wmma_f32_16x16x32_f16(false, a3, false, bf3, (short)0, c1, false, false);

            // retire this 16x16 tile immediately: bias2 + silu + W3 dot
            const int n   = nt * 16 + row;
            const float bn = sB2[n];
            const float wn = sW3[n];
            float h[8], sg[8];
            #pragma unroll
            for (int v = 0; v < 8; ++v) h[v] = (c0[v] + c1[v]) + bn;
            #pragma unroll
            for (int v = 0; v < 8; ++v) sg[v] = __expf(-h[v]);
            #pragma unroll
            for (int v = 0; v < 8; ++v) sg[v] = __builtin_amdgcn_rcpf(1.0f + sg[v]);
            #pragma unroll
            for (int v = 0; v < 8; ++v) acc += wn * (h[v] * sg[v]);
        }

        // wave32 reduce
        #pragma unroll
        for (int off = 16; off >= 1; off >>= 1)
            acc += __shfl_xor(acc, off, 32);

        if (lane == 0) atomicAdd(&out[b], 0.5f * acc);
    }
}

extern "C" void kernel_launch(void* const* d_in, const int* in_sizes, int n_in,
                              void* d_out, int out_size, void* d_ws, size_t ws_size,
                              hipStream_t stream) {
    const float* pos = (const float*)d_in[0];
    const float* W1  = (const float*)d_in[1];
    const float* b1  = (const float*)d_in[2];
    const float* W2  = (const float*)d_in[3];
    const float* b2  = (const float*)d_in[4];
    const float* W3  = (const float*)d_in[5];
    const float* b3  = (const float*)d_in[6];
    float* out = (float*)d_out;

    disc_init_out<<<1, 32, 0, stream>>>(out, b3);
    disc_pair_mlp<<<1024, 256, 0, stream>>>(pos, W1, b1, W2, b2, W3, out);
}
